// SAMoELayer_55688545960244
// MI455X (gfx1250) — compile-verified, matrix operand order checked by reading, not run on hardware
//
#include <hip/hip_runtime.h>
#include <math.h>

#define Dm   1024
#define Em   8
#define FFm  4096
#define Bm   2
#define Tm   2048
#define NTOK (Bm * Tm)   // 4096
#define SEm  64
#define SHm  128

typedef __attribute__((ext_vector_type(16))) __bf16 v16bf;
typedef __attribute__((ext_vector_type(8)))  float  v8f;

// RNE fp32->bf16 (small activation volumes)
__device__ __forceinline__ unsigned short f2bf(float f) {
    unsigned int u = __float_as_uint(f);
    unsigned int r = u + 0x7FFFu + ((u >> 16) & 1u);
    return (unsigned short)(r >> 16);
}

// pack {bf16(lo), bf16(hi)} into one dword in ONE VALU op (bulk weight path)
#if __has_builtin(__builtin_amdgcn_cvt_pk_bf16_f32)
typedef __attribute__((ext_vector_type(2))) __bf16 v2bf;
__device__ __forceinline__ unsigned int pk2(float lo, float hi) {
    union { v2bf h; unsigned int u; } r;
    r.h = __builtin_amdgcn_cvt_pk_bf16_f32(lo, hi);   // v_cvt_pk_bf16_f32 (RNE)
    return r.u;
}
#else
__device__ __forceinline__ unsigned int pk2(float lo, float hi) {
    // v_perm_b32: take high halves (truncate-to-bf16)
    return __builtin_amdgcn_perm(__float_as_uint(hi), __float_as_uint(lo), 0x07060302u);
}
#endif

#if __has_builtin(__builtin_amdgcn_global_load_async_to_lds_b128) && \
    __has_builtin(__builtin_amdgcn_s_wait_asynccnt)
#define HAVE_ASYNC_LDS 1
typedef int v4i __attribute__((vector_size(16)));
typedef __attribute__((address_space(3))) v4i* lds_v4i_p;
#endif

__device__ __forceinline__ void atomAddF(float* p, float v) {
    __hip_atomic_fetch_add(p, v, __ATOMIC_RELAXED, __HIP_MEMORY_SCOPE_AGENT);
}

// ---------------------------------------------------------------- init: out = x, cnt = 0
__global__ void init_kernel(const float4* __restrict__ x, float4* __restrict__ out,
                            int* __restrict__ cnt, int total4) {
    int i = blockIdx.x * blockDim.x + threadIdx.x;
    if (i < total4) out[i] = x[i];
    if (blockIdx.x == 0 && threadIdx.x < Em) cnt[threadIdx.x] = 0;
}

// ---------------------------------------------------------------- LayerNorm over D
__global__ void ln_kernel(const float* __restrict__ x, const float* __restrict__ g,
                          const float* __restrict__ bta, float* __restrict__ h_ln) {
    int n = blockIdx.x, tid = threadIdx.x;
    __shared__ float rs[256], rs2[256];
    const float* xp = x + (size_t)n * Dm;
    float v[4]; float s = 0.f, s2 = 0.f;
#pragma unroll
    for (int i = 0; i < 4; i++) { v[i] = xp[tid + i * 256]; s += v[i]; s2 += v[i] * v[i]; }
    rs[tid] = s; rs2[tid] = s2; __syncthreads();
    for (int st = 128; st > 0; st >>= 1) {
        if (tid < st) { rs[tid] += rs[tid + st]; rs2[tid] += rs2[tid + st]; }
        __syncthreads();
    }
    float mu  = rs[0] * (1.0f / Dm);
    float var = rs2[0] * (1.0f / Dm) - mu * mu;
    float inv = rsqrtf(var + 1e-5f);
    float* hp = h_ln + (size_t)n * Dm;
#pragma unroll
    for (int i = 0; i < 4; i++) { int d = tid + i * 256; hp[d] = (v[i] - mu) * inv * g[d] + bta[d]; }
}

// ---------------------------------------------------------------- per-(b,d) stats over T
__global__ void tstats_kernel(const float* __restrict__ h_ln,
                              float* __restrict__ m_t, float* __restrict__ rs_t) {
    int idx = blockIdx.x * blockDim.x + threadIdx.x;
    if (idx >= Bm * Dm) return;
    int b = idx / Dm, d = idx % Dm;
    const float* p = h_ln + (size_t)b * Tm * Dm + d;
    float s = 0.f, s2 = 0.f;
    for (int t = 0; t < Tm; t++) { float v = p[(size_t)t * Dm]; s += v; s2 += v * v; }
    float mu = s * (1.f / Tm), var = s2 * (1.f / Tm) - mu * mu;
    m_t[idx]  = mu;
    rs_t[idx] = rsqrtf(var + 1e-8f);
}

// ---------------------------------------------------------------- style hypernetwork
__global__ void hyper_kernel(const int* __restrict__ sid, const float* __restrict__ subj_emb,
                             const float* __restrict__ w1, const float* __restrict__ b1,
                             const float* __restrict__ w2, const float* __restrict__ b2,
                             const float* __restrict__ sw1, const float* __restrict__ sb1,
                             const float* __restrict__ sw2, const float* __restrict__ sb2,
                             float* __restrict__ gam, float* __restrict__ bet) {
    int b = blockIdx.x, tid = threadIdx.x;      // 128 threads
    __shared__ float se[SEm], h1[SHm], h2[SHm], s1[SEm];
    if (tid < SEm) se[tid] = subj_emb[(size_t)sid[b] * SEm + tid];
    __syncthreads();
    { float a = b1[tid]; for (int i = 0; i < SEm; i++) a += se[i] * w1[i * SHm + tid]; h1[tid] = fmaxf(a, 0.f); }
    __syncthreads();
    { float a = b2[tid]; for (int i = 0; i < SHm; i++) a += h1[i] * w2[i * SHm + tid]; h2[tid] = a; }
    __syncthreads();
    if (tid < SEm) { float a = sb1[tid]; for (int i = 0; i < SHm; i++) a += h2[i] * sw1[i * SEm + tid]; s1[tid] = fmaxf(a, 0.f); }
    __syncthreads();
    for (int j = tid; j < 2 * Dm; j += SHm) {
        float a = sb2[j];
        for (int i = 0; i < SEm; i++) a += s1[i] * sw2[(size_t)i * 2 * Dm + j];
        if (j < Dm) {
            float sp = (a > 20.f) ? a : log1pf(__expf(a));   // softplus
            gam[(size_t)b * Dm + j] = sp + 1e-8f;
        } else {
            bet[(size_t)b * Dm + (j - Dm)] = a;
        }
    }
}

// ---------------------------------------------------------------- modulate + router + dispatch lists
__global__ void route_mod_kernel(const float* __restrict__ h_ln, const float* __restrict__ m_t,
                                 const float* __restrict__ rs_t, const float* __restrict__ gam,
                                 const float* __restrict__ bet, const float* __restrict__ router_w,
                                 unsigned short* __restrict__ h_bf, int* __restrict__ cnt,
                                 int* __restrict__ tokl, float* __restrict__ gatel) {
    int n = blockIdx.x, tid = threadIdx.x;
    int b = n / Tm;
    __shared__ float lred[256 * Em];
    float part[Em];
#pragma unroll
    for (int e = 0; e < Em; e++) part[e] = 0.f;
    const float* hp = h_ln + (size_t)n * Dm;
    const float* mp = m_t  + (size_t)b * Dm;
    const float* rp = rs_t + (size_t)b * Dm;
    const float* gp = gam  + (size_t)b * Dm;
    const float* bp = bet  + (size_t)b * Dm;
    unsigned short* hb = h_bf + (size_t)n * Dm;
    for (int d = tid; d < Dm; d += 256) {
        float hv = (hp[d] - mp[d]) * rp[d] * gp[d] + bp[d];
        hb[d] = f2bf(hv);
#pragma unroll
        for (int e = 0; e < Em; e++) part[e] += hv * router_w[d * Em + e];
    }
#pragma unroll
    for (int e = 0; e < Em; e++) lred[tid * Em + e] = part[e];
    __syncthreads();
    for (int st = 128; st > 0; st >>= 1) {
        if (tid < st) {
#pragma unroll
            for (int e = 0; e < Em; e++) lred[tid * Em + e] += lred[(tid + st) * Em + e];
        }
        __syncthreads();
    }
    if (tid == 0) {
        float p[Em];
        float mx = lred[0];
        for (int e = 1; e < Em; e++) mx = fmaxf(mx, lred[e]);
        for (int e = 0; e < Em; e++) p[e] = __expf(lred[e] - mx);
        int i0 = 0;
        for (int e = 1; e < Em; e++) if (p[e] > p[i0]) i0 = e;
        int i1 = (i0 == 0) ? 1 : 0;
        for (int e = 0; e < Em; e++) if (e != i0 && p[e] > p[i1]) i1 = e;
        float g0 = p[i0], g1 = p[i1], gs = g0 + g1;   // renorm over top-2
        g0 /= gs; g1 /= gs;
        int p0 = atomicAdd(&cnt[i0], 1); tokl[i0 * NTOK + p0] = n; gatel[i0 * NTOK + p0] = g0;
        int p1 = atomicAdd(&cnt[i1], 1); tokl[i1 * NTOK + p1] = n; gatel[i1 * NTOK + p1] = g1;
    }
}

// ---------------------------------------------------------------- expert FFN (bf16 WMMA, LDS-staged weights)
// Block = 16 gathered tokens of one expert, 8 wave32s.
// Per 128-wide FF chunk:
//   GEMM1: hid(16x128) = relu(X(16x1024) @ W1chunk + b1); W1 staged in 64k x 128f slabs, transposed [f][k].
//   GEMM2: out(16x1024) += hid @ W2chunk; W2 staged in 32k x 256d slabs, transposed [d][k].
// B fragments = two ds_load_b128 per lane (16-bit layout: K = {0..7,16..23} + 8*(lane>=16)).
__global__ __launch_bounds__(256) void expert_kernel(
    const unsigned short* __restrict__ h_bf, const int* __restrict__ cnt,
    const int* __restrict__ tokl, const float* __restrict__ gatel,
    const float* __restrict__ w1g, const float* __restrict__ b1g,
    const float* __restrict__ w2g, const float* __restrict__ b2g,
    float* __restrict__ out) {
    int e  = blockIdx.y;
    int c  = cnt[e];
    int t0 = blockIdx.x * 16;
    if (t0 >= c) return;                        // block-uniform: EXEC all-ones for WMMA
    int tid = threadIdx.x, lane = tid & 31, wave = tid >> 5;

    __shared__ __attribute__((aligned(16))) unsigned short Xs[16][1032];  // 33.0 KB
    __shared__ __attribute__((aligned(16))) unsigned short Hs[16][136];   //  4.4 KB
    __shared__ __attribute__((aligned(16))) unsigned short Bst[10240];    // 20.5 KB (reused W1/W2 stage)
    __shared__ int   toks[16];
    __shared__ float gts[16];

    if (tid < 16) {
        int p = t0 + tid;
        toks[tid] = (p < c) ? tokl[e * NTOK + p] : 0;
        gts[tid]  = (p < c) ? gatel[e * NTOK + p] : 0.f;   // pad rows contribute 0
    }
    __syncthreads();
    {   // stage X tile: 16 tokens x 1024 bf16
        int row = tid >> 4, cb = (tid & 15) * 64;
        const unsigned short* srcg = h_bf + (size_t)toks[row] * Dm + cb;
        unsigned short* dstl = &Xs[row][cb];
#ifdef HAVE_ASYNC_LDS
#pragma unroll
        for (int i = 0; i < 8; i++)
            __builtin_amdgcn_global_load_async_to_lds_b128(
                (v4i*)(srcg + i * 8), (lds_v4i_p)(dstl + i * 8), 0, 0);
        __builtin_amdgcn_s_wait_asynccnt(0);
#else
        const uint4* src = (const uint4*)srcg;
        uint4* dst = (uint4*)dstl;
#pragma unroll
        for (int i = 0; i < 8; i++) dst[i] = src[i];
#endif
    }

    const float* W1 = w1g + (size_t)e * Dm * FFm;   // [k][f], stride FFm
    const float* W2 = w2g + (size_t)e * FFm * Dm;   // [f][d], stride Dm

    v8f acc2[8];
#pragma unroll
    for (int t = 0; t < 8; t++)
#pragma unroll
        for (int r = 0; r < 8; r++) acc2[t][r] = 0.f;

    int m  = lane & 15;               // matrix row / B-col lane index
    int hk = (lane >> 4) << 3;        // +8 K-offset for upper half-wave

    for (int fc = 0; fc < FFm; fc += 128) {
        // ======== GEMM1 ========
        v8f acc1;
#pragma unroll
        for (int r = 0; r < 8; r++) acc1[r] = 0.f;
        int fcol = fc + wave * 16 + m;
        __builtin_prefetch(&W1[(size_t)(fc + 128)], 0, 1);
        for (int k0 = 0; k0 < Dm; k0 += 64) {
            __syncthreads();        // Bst consumers of previous slab done
            // stage W1[k0..k0+64][fc..fc+128] -> Bst[f(128)][k(64)] stride 72, bf16
#pragma unroll
            for (int q = 0; q < 4; q++) {
                int p  = tid + q * 256;        // 0..1023
                int k2 = (p >> 5) << 1;        // 0,2,...,62
                int fq = (p & 31) << 2;        // 0,4,...,124
                const float* s0 = W1 + (size_t)(k0 + k2) * FFm + (fc + fq);
                float4 a = *(const float4*)s0;
                float4 b = *(const float4*)(s0 + FFm);
                *(unsigned int*)&Bst[(fq + 0) * 72 + k2] = pk2(a.x, b.x);
                *(unsigned int*)&Bst[(fq + 1) * 72 + k2] = pk2(a.y, b.y);
                *(unsigned int*)&Bst[(fq + 2) * 72 + k2] = pk2(a.z, b.z);
                *(unsigned int*)&Bst[(fq + 3) * 72 + k2] = pk2(a.w, b.w);
            }
            __syncthreads();
            int fl = wave * 16 + m;
#pragma unroll
            for (int kk = 0; kk < 64; kk += 32) {
                union { uint4 q[2]; v16bf v; } A;
                A.q[0] = *(const uint4*)&Xs[m][k0 + kk + hk];
                A.q[1] = *(const uint4*)&Xs[m][k0 + kk + hk + 16];
                union { uint4 q[2]; v16bf v; } Bf;
                Bf.q[0] = *(const uint4*)&Bst[fl * 72 + kk + hk];
                Bf.q[1] = *(const uint4*)&Bst[fl * 72 + kk + hk + 16];
                acc1 = __builtin_amdgcn_wmma_f32_16x16x32_bf16(false, A.v, false, Bf.v,
                                                               (short)0, acc1, false, false);
            }
        }
        float bb = b1g[(size_t)e * FFm + fcol];
#pragma unroll
        for (int r = 0; r < 8; r++) {
            float vv = fmaxf(acc1[r] + bb, 0.f);
            Hs[r + hk][wave * 16 + m] = f2bf(vv);
        }
        __syncthreads();            // all hid columns visible before A2 reads
        // ======== GEMM2 ========
        __builtin_prefetch(&W2[(size_t)(fc + 128) * Dm], 0, 1);
        for (int ks = 0; ks < 128; ks += 32) {
            union { uint4 q[2]; v16bf v; } A2;
            A2.q[0] = *(const uint4*)&Hs[m][ks + hk];
            A2.q[1] = *(const uint4*)&Hs[m][ks + hk + 16];
            for (int s = 0; s < 4; s++) {
                __syncthreads();    // Bst consumers of previous stage done
                // stage W2[fc+ks .. +32][s*256 .. +256] -> Bst[d(256)][k(32)] stride 40, bf16
#pragma unroll
                for (int q = 0; q < 4; q++) {
                    int p  = tid + q * 256;        // 0..1023
                    int k2 = (p >> 6) << 1;        // 0,2,...,30
                    int dq = (p & 63) << 2;        // 0,4,...,252
                    const float* s0 = W2 + (size_t)(fc + ks + k2) * Dm + s * 256 + dq;
                    float4 a = *(const float4*)s0;
                    float4 b = *(const float4*)(s0 + Dm);
                    *(unsigned int*)&Bst[(dq + 0) * 40 + k2] = pk2(a.x, b.x);
                    *(unsigned int*)&Bst[(dq + 1) * 40 + k2] = pk2(a.y, b.y);
                    *(unsigned int*)&Bst[(dq + 2) * 40 + k2] = pk2(a.z, b.z);
                    *(unsigned int*)&Bst[(dq + 3) * 40 + k2] = pk2(a.w, b.w);
                }
                __syncthreads();
#pragma unroll
                for (int t2 = 0; t2 < 2; t2++) {
                    int dl = (wave * 2 + t2) * 16 + m;
                    union { uint4 q[2]; v16bf v; } Bf;
                    Bf.q[0] = *(const uint4*)&Bst[dl * 40 + hk];
                    Bf.q[1] = *(const uint4*)&Bst[dl * 40 + hk + 16];
                    acc2[s * 2 + t2] = __builtin_amdgcn_wmma_f32_16x16x32_bf16(
                        false, A2.v, false, Bf.v, (short)0, acc2[s * 2 + t2], false, false);
                }
            }
        }
    }
    // epilogue: y += gate * (acc + b2)   (gates sum to 1 across top-2 -> b2 exact)
#pragma unroll
    for (int s = 0; s < 4; s++)
#pragma unroll
        for (int t2 = 0; t2 < 2; t2++) {
            int d = s * 256 + (wave * 2 + t2) * 16 + m;
            float b2v = b2g[(size_t)e * Dm + d];
#pragma unroll
            for (int r = 0; r < 8; r++) {
                int mr = r + hk;
                float val = (acc2[s * 2 + t2][r] + b2v) * gts[mr];
                atomAddF(&out[(size_t)toks[mr] * Dm + d], val);
            }
        }
}

// ---------------------------------------------------------------- host launch
extern "C" void kernel_launch(void* const* d_in, const int* in_sizes, int n_in,
                              void* d_out, int out_size, void* d_ws, size_t ws_size,
                              hipStream_t stream) {
    (void)in_sizes; (void)n_in; (void)out_size; (void)ws_size;
    const float* x        = (const float*)d_in[0];
    const int*   sid      = (const int*)  d_in[1];
    const float* ln_g     = (const float*)d_in[2];
    const float* ln_b     = (const float*)d_in[3];
    const float* subj_emb = (const float*)d_in[4];
    const float* hyp_w1   = (const float*)d_in[5];
    const float* hyp_b1   = (const float*)d_in[6];
    const float* hyp_w2   = (const float*)d_in[7];
    const float* hyp_b2   = (const float*)d_in[8];
    const float* sty_w1   = (const float*)d_in[9];
    const float* sty_b1   = (const float*)d_in[10];
    const float* sty_w2   = (const float*)d_in[11];
    const float* sty_b2   = (const float*)d_in[12];
    const float* router_w = (const float*)d_in[13];
    const float* exp_w1   = (const float*)d_in[14];
    const float* exp_b1   = (const float*)d_in[15];
    const float* exp_w2   = (const float*)d_in[16];
    const float* exp_b2   = (const float*)d_in[17];
    float* out = (float*)d_out;

    char* w = (char*)d_ws;
    float* h_ln = (float*)w;                    w += (size_t)NTOK * Dm * sizeof(float);
    unsigned short* h_bf = (unsigned short*)w;  w += (size_t)NTOK * Dm * sizeof(unsigned short);
    float* m_t  = (float*)w;                    w += (size_t)Bm * Dm * sizeof(float);
    float* rs_t = (float*)w;                    w += (size_t)Bm * Dm * sizeof(float);
    float* gam  = (float*)w;                    w += (size_t)Bm * Dm * sizeof(float);
    float* bet  = (float*)w;                    w += (size_t)Bm * Dm * sizeof(float);
    int*   cnt  = (int*)w;                      w += 64;
    int*   tokl = (int*)w;                      w += (size_t)Em * NTOK * sizeof(int);
    float* gatel = (float*)w;                   w += (size_t)Em * NTOK * sizeof(float);

    int total4 = NTOK * Dm / 4;
    init_kernel<<<(total4 + 255) / 256, 256, 0, stream>>>((const float4*)x, (float4*)out, cnt, total4);
    ln_kernel<<<NTOK, 256, 0, stream>>>(x, ln_g, ln_b, h_ln);
    tstats_kernel<<<(Bm * Dm) / 256, 256, 0, stream>>>(h_ln, m_t, rs_t);
    hyper_kernel<<<Bm, SHm, 0, stream>>>(sid, subj_emb, hyp_w1, hyp_b1, hyp_w2, hyp_b2,
                                         sty_w1, sty_b1, sty_w2, sty_b2, gam, bet);
    route_mod_kernel<<<NTOK, 256, 0, stream>>>(h_ln, m_t, rs_t, gam, bet, router_w,
                                               h_bf, cnt, tokl, gatel);
    expert_kernel<<<dim3(NTOK / 16, Em), 256, 0, stream>>>(h_bf, cnt, tokl, gatel,
                                                           exp_w1, exp_b1, exp_w2, exp_b2, out);
}
